// SparseAutoencoderComplete_66812511256588
// MI455X (gfx1250) — compile-verified, matrix operand order checked by reading, not run on hardware
//
#include <hip/hip_runtime.h>
#include <hip/hip_bf16.h>

typedef __attribute__((ext_vector_type(16))) __bf16         v16bf;
typedef __attribute__((ext_vector_type(8)))  __bf16         v8bf;
typedef __attribute__((ext_vector_type(8)))  unsigned short v8us;
typedef __attribute__((ext_vector_type(8)))  float          v8f;
typedef __attribute__((ext_vector_type(4)))  float          v4f;

#define BATCH   131072
#define INSZ    784
#define HID     64
#define KTOP    20
#define KCH_ENC 25   // ceil(784/32), last chunk half-padded
#define NT_ENC  4    // 64/16
#define KCH_DEC 2    // 64/32
#define NT_DEC  49   // 784/16

#define NENC (KCH_ENC * NT_ENC * 32 * 16)   // 51200 bf16 elems
#define NDEC (NT_DEC * KCH_DEC * 32 * 16)   // 50176 bf16 elems

__device__ __forceinline__ unsigned short f2bf_bits(float f) {
    __bf16 b = (__bf16)f;
    return __builtin_bit_cast(unsigned short, b);
}

// ---------------------------------------------------------------------------
// Prep: convert f32 weights into bf16 B-fragments pre-swizzled to the CDNA5
// 16-bit WMMA operand layout.  Fragment element e (0..15) of lane L maps to
// K = (e>>3)*16 + (L>>4)*8 + (e&7), column N = L&15.
// Encoder B[k][n] = W_enc[16t+n][k]; Decoder B[k][n] = W_dec[16nt+n][k].
// ---------------------------------------------------------------------------
__global__ void sae_prep_weights(const float* __restrict__ W_enc,
                                 const float* __restrict__ W_dec,
                                 unsigned short* __restrict__ benc,
                                 unsigned short* __restrict__ bdec) {
    int id = blockIdx.x * 256 + threadIdx.x;
    if (id < NENC) {
        int e = id & 15, lane = (id >> 4) & 31, t = (id >> 9) & 3, c = id >> 11;
        int sel = lane >> 4, n = lane & 15, d = e & 7, grp = e >> 3;
        int k = c * 32 + grp * 16 + sel * 8 + d;
        float v = (k < INSZ) ? W_enc[(t * 16 + n) * INSZ + k] : 0.0f;
        benc[id] = f2bf_bits(v);
    } else if (id < NENC + NDEC) {
        int id2 = id - NENC;
        int e = id2 & 15, lane = (id2 >> 4) & 31, c = (id2 >> 9) & 1, nt = id2 >> 10;
        int sel = lane >> 4, n = lane & 15, d = e & 7, grp = e >> 3;
        int k = c * 32 + grp * 16 + sel * 8 + d;                 // < 64
        float v = W_dec[(nt * 16 + n) * HID + k];
        bdec[id2] = f2bf_bits(v);
    }
}

// ---------------------------------------------------------------------------
// Main: 8 wave32 per block, each wave owns 16 rows of x.
// Streaming traffic (x in, h / x_hat out) uses non-temporal TH hints so the
// high-reuse bf16 weight fragments stay resident in WGP$/L2.
// ---------------------------------------------------------------------------
__global__ __launch_bounds__(256) void sae_fwd(
    const float* __restrict__ x,
    const float* __restrict__ b_enc,
    const float* __restrict__ b_dec,
    const unsigned short* __restrict__ benc_us,
    const unsigned short* __restrict__ bdec_us,
    float* __restrict__ h_out,
    float* __restrict__ xhat_out)
{
    __shared__ __align__(16) unsigned short lds_h[8][16 * HID];   // 16 KB

    const int lane = threadIdx.x & 31;
    const int wave = threadIdx.x >> 5;
    const int sel  = lane >> 4;        // K-half select (A & B 16-bit layout)
    const int n16  = lane & 15;        // row (A) / column (B,C) within tile
    const int rowBase = blockIdx.x * 128 + wave * 16;

    const v16bf* bencv = (const v16bf*)benc_us;
    const v16bf* bdecv = (const v16bf*)bdec_us;
    const float* xrow  = x + (size_t)(rowBase + n16) * INSZ;

    // ---------------- encoder GEMM: pre[16x64] = x_tile @ W_enc^T ----------
    v8f acc0 = {0,0,0,0,0,0,0,0}, acc1 = {0,0,0,0,0,0,0,0};
    v8f acc2 = {0,0,0,0,0,0,0,0}, acc3 = {0,0,0,0,0,0,0,0};

    #pragma unroll
    for (int c = 0; c < KCH_ENC; ++c) {
        const float* xp = xrow + c * 32 + sel * 8;
        if (c < KCH_ENC - 2) __builtin_prefetch(xp + 32, 0, 1);
        v4f f0 = __builtin_nontemporal_load((const v4f*)(xp));
        v4f f1 = __builtin_nontemporal_load((const v4f*)(xp + 4));
        v16bf a;
        a[0]=(__bf16)f0[0]; a[1]=(__bf16)f0[1]; a[2]=(__bf16)f0[2]; a[3]=(__bf16)f0[3];
        a[4]=(__bf16)f1[0]; a[5]=(__bf16)f1[1]; a[6]=(__bf16)f1[2]; a[7]=(__bf16)f1[3];
        if (c < KCH_ENC - 1) {
            v4f f2 = __builtin_nontemporal_load((const v4f*)(xp + 16));
            v4f f3 = __builtin_nontemporal_load((const v4f*)(xp + 20));
            a[8]=(__bf16)f2[0];  a[9]=(__bf16)f2[1];  a[10]=(__bf16)f2[2]; a[11]=(__bf16)f2[3];
            a[12]=(__bf16)f3[0]; a[13]=(__bf16)f3[1]; a[14]=(__bf16)f3[2]; a[15]=(__bf16)f3[3];
        } else {
            // K tail: k = 784..799 -> zero pad
            #pragma unroll
            for (int i = 8; i < 16; ++i) a[i] = (__bf16)0.0f;
        }
        acc0 = __builtin_amdgcn_wmma_f32_16x16x32_bf16(false, a, false, bencv[(c*4+0)*32+lane], (short)0, acc0, false, false);
        acc1 = __builtin_amdgcn_wmma_f32_16x16x32_bf16(false, a, false, bencv[(c*4+1)*32+lane], (short)0, acc1, false, false);
        acc2 = __builtin_amdgcn_wmma_f32_16x16x32_bf16(false, a, false, bencv[(c*4+2)*32+lane], (short)0, acc2, false, false);
        acc3 = __builtin_amdgcn_wmma_f32_16x16x32_bf16(false, a, false, bencv[(c*4+3)*32+lane], (short)0, acc3, false, false);
    }

    // ---------------- bias + ReLU (f32) ------------------------------------
    // C layout: acc_t[v] holds pre[M = v + 8*sel][N = 16t + n16]
    const float be0 = b_enc[n16],      be1 = b_enc[16 + n16];
    const float be2 = b_enc[32 + n16], be3 = b_enc[48 + n16];
    float p0[8], p1[8], p2[8], p3[8];
    #pragma unroll
    for (int v = 0; v < 8; ++v) {
        p0[v] = fmaxf(acc0[v] + be0, 0.0f);
        p1[v] = fmaxf(acc1[v] + be1, 0.0f);
        p2[v] = fmaxf(acc2[v] + be2, 0.0f);
        p3[v] = fmaxf(acc3[v] + be3, 0.0f);
    }

    // ---------------- top-20 threshold per row -----------------------------
    // Values >= 0 -> IEEE bit order == numeric order. Binary-search the
    // largest T with count(bits >= T) >= 20; row spans 16 lanes (4 vals each),
    // counts reduced with xor-shuffles confined to each 16-lane half.
    unsigned thr[8] = {0,0,0,0,0,0,0,0};
    for (int bit = 30; bit >= 0; --bit) {
        const unsigned bm = 1u << bit;
        #pragma unroll
        for (int v = 0; v < 8; ++v) {
            unsigned cand = thr[v] | bm;
            int cnt = (int)(__float_as_uint(p0[v]) >= cand)
                    + (int)(__float_as_uint(p1[v]) >= cand)
                    + (int)(__float_as_uint(p2[v]) >= cand)
                    + (int)(__float_as_uint(p3[v]) >= cand);
            cnt += __shfl_xor(cnt, 1, 32);
            cnt += __shfl_xor(cnt, 2, 32);
            cnt += __shfl_xor(cnt, 4, 32);
            cnt += __shfl_xor(cnt, 8, 32);
            thr[v] = (cnt >= KTOP) ? cand : thr[v];
        }
    }

    // ---------------- gate, emit h (f32 global NT + bf16 LDS) --------------
    float* hb = h_out + (size_t)rowBase * HID;
    #pragma unroll
    for (int v = 0; v < 8; ++v) {
        const int m = v + 8 * sel;
        float q0 = (__float_as_uint(p0[v]) >= thr[v]) ? p0[v] : 0.0f;
        float q1 = (__float_as_uint(p1[v]) >= thr[v]) ? p1[v] : 0.0f;
        float q2 = (__float_as_uint(p2[v]) >= thr[v]) ? p2[v] : 0.0f;
        float q3 = (__float_as_uint(p3[v]) >= thr[v]) ? p3[v] : 0.0f;
        __builtin_nontemporal_store(q0, &hb[(size_t)m * HID +      n16]);
        __builtin_nontemporal_store(q1, &hb[(size_t)m * HID + 16 + n16]);
        __builtin_nontemporal_store(q2, &hb[(size_t)m * HID + 32 + n16]);
        __builtin_nontemporal_store(q3, &hb[(size_t)m * HID + 48 + n16]);
        lds_h[wave][m * HID +      n16] = f2bf_bits(q0);
        lds_h[wave][m * HID + 16 + n16] = f2bf_bits(q1);
        lds_h[wave][m * HID + 32 + n16] = f2bf_bits(q2);
        lds_h[wave][m * HID + 48 + n16] = f2bf_bits(q3);
    }
    __syncthreads();

    // ---------------- decoder A fragments from LDS (transpose of C layout) -
    const unsigned short* hl = &lds_h[wave][n16 * HID];
    v8us u0 = *(const v8us*)(hl +      sel * 8);   // k =      sel*8 + 0..7
    v8us u1 = *(const v8us*)(hl + 16 + sel * 8);   // k = 16 + sel*8 + 0..7
    v8us u2 = *(const v8us*)(hl + 32 + sel * 8);   // k = 32 + sel*8 + 0..7
    v8us u3 = *(const v8us*)(hl + 48 + sel * 8);   // k = 48 + sel*8 + 0..7
    v16bf af0 = __builtin_shufflevector(__builtin_bit_cast(v8bf, u0),
                                        __builtin_bit_cast(v8bf, u1),
                                        0,1,2,3,4,5,6,7,8,9,10,11,12,13,14,15);
    v16bf af1 = __builtin_shufflevector(__builtin_bit_cast(v8bf, u2),
                                        __builtin_bit_cast(v8bf, u3),
                                        0,1,2,3,4,5,6,7,8,9,10,11,12,13,14,15);

    // ---------------- decoder GEMM: x_hat[16x784] = h @ W_dec^T + b_dec ----
    float* xr = xhat_out + (size_t)rowBase * INSZ;
    for (int nt = 0; nt < NT_DEC; ++nt) {
        v8f cc = {0,0,0,0,0,0,0,0};
        cc = __builtin_amdgcn_wmma_f32_16x16x32_bf16(false, af0, false, bdecv[(nt*2+0)*32+lane], (short)0, cc, false, false);
        cc = __builtin_amdgcn_wmma_f32_16x16x32_bf16(false, af1, false, bdecv[(nt*2+1)*32+lane], (short)0, cc, false, false);
        const float bd = b_dec[nt * 16 + n16];
        #pragma unroll
        for (int v = 0; v < 8; ++v) {
            __builtin_nontemporal_store(cc[v] + bd,
                &xr[(size_t)(v + 8 * sel) * INSZ + nt * 16 + n16]);
        }
    }
}

extern "C" void kernel_launch(void* const* d_in, const int* in_sizes, int n_in,
                              void* d_out, int out_size, void* d_ws, size_t ws_size,
                              hipStream_t stream) {
    (void)in_sizes; (void)n_in; (void)out_size; (void)ws_size;
    const float* x     = (const float*)d_in[0];
    const float* W_enc = (const float*)d_in[1];
    const float* b_enc = (const float*)d_in[2];
    const float* W_dec = (const float*)d_in[3];
    const float* b_dec = (const float*)d_in[4];

    unsigned short* benc = (unsigned short*)d_ws;           // 102400 B
    unsigned short* bdec = benc + NENC;                     // 100352 B

    float* h_out = (float*)d_out;                           // [B, 64]
    float* xhat  = h_out + (size_t)BATCH * HID;             // [B, 784]

    const int prep_threads = NENC + NDEC;                   // 101376
    sae_prep_weights<<<dim3((prep_threads + 255) / 256), dim3(256), 0, stream>>>(
        W_enc, W_dec, benc, bdec);

    sae_fwd<<<dim3(BATCH / 128), dim3(256), 0, stream>>>(
        x, b_enc, b_dec, benc, bdec, h_out, xhat);
}
